// EncoderLayer_29970281792187
// MI455X (gfx1250) — matrix operand order repaired and checked by reading, not compiled, not run on hardware
//
#include <hip/hip_runtime.h>
#include <stdint.h>

#define H    256
#define G3   768
#define TT   512
#define NSEQ 256
#define MB   16            // sequences per workgroup (16 -> 16 WGs: 2x step parallelism)
#define NWG  (NSEQ / MB)   // 16
#define ASTR 536           // padded K-stride of A panel (512 + pad, breaks LDS bank conflicts)
#define P1_ITERS (MB / 4)  // phase-1 tiles per wave: (MB/16)*32 / 8
#define P2_ITERS (MB / 8)  // phase-2 tiles per wave: (MB/16)*16 / 8
#define PACK_ELEMS (48 * 8 * 32 * 16)  // packed bf16 elements per 768x256 matrix

typedef __attribute__((ext_vector_type(16))) __bf16 v16bf;
typedef __attribute__((ext_vector_type(8)))  float  v8f;

union BfFrag { v16bf v; uint4 q[2]; };

__device__ __forceinline__ unsigned short f2bf(float f) {
  unsigned int u = __float_as_uint(f);
  unsigned int r = u + 0x7FFFu + ((u >> 16) & 1u);   // round-to-nearest-even
  return (unsigned short)(r >> 16);
}
__device__ __forceinline__ float bf2f(unsigned short s) {
  return __uint_as_float(((unsigned int)s) << 16);
}
__device__ __forceinline__ float sigmoidf(float x) {
  return 1.0f / (1.0f + __expf(-x));
}

// Pack a [768,256] row-major f32 weight matrix (gh = h @ W^T) into bf16
// WMMA B-fragments: B[k][n] with per-WMMA-frag layout
//   out[(((c*8)+kf)*32 + lane)*16 + j] = W[c*16 + (lane&15)][kf*32 + 8*(lane>>4) + (j&7) + 16*(j>>3)]
__global__ void gru_pack_b(const float* __restrict__ W, unsigned short* __restrict__ out) {
  int i = blockIdx.x * blockDim.x + threadIdx.x;
  if (i >= PACK_ELEMS) return;
  int j    =  i        & 15;
  int lane = (i >> 4)  & 31;
  int kf   = (i >> 9)  & 7;
  int c    =  i >> 12;
  out[i] = f2bf(W[(c * 16 + (lane & 15)) * H + (kf * 32 + ((lane >> 4) << 3) + (j & 7) + ((j >> 3) << 4))]);
}

__global__ __launch_bounds__(256)
void gru_layer_kernel(int layer,
                      const float* __restrict__ x_in,           // L0: inputs [8,512,32]; else y_prev [256,512,256]
                      const unsigned short* __restrict__ whh_p, // packed bf16 B frags
                      const unsigned short* __restrict__ wih_p, // packed bf16 B frags (unused for L0)
                      const float* __restrict__ wih0,           // [768] raw W_ih for L0
                      const float* __restrict__ bih,
                      const float* __restrict__ bhh,
                      float* __restrict__ y_out)
{
  __shared__ __align__(16) unsigned short A[MB * ASTR]; // [row][k] bf16; k<256: h, 256..511: x_t
  __shared__ unsigned short r_s[MB * H];
  __shared__ unsigned short z_s[MB * H];
  __shared__ unsigned short hn_s[MB * H];
  __shared__ float bih_s[G3];
  __shared__ float bhh_s[G3];
  __shared__ float wih0_s[G3];
  __shared__ float xs[MB];

  const int tid  = threadIdx.x;
  const int lane = tid & 31;
  const int w    = tid >> 5;            // wave id 0..7
  const int base = blockIdx.x * MB;
  const int m_hi = lane >> 4;           // 0/1: which K-subgroup of the A frag
  const int nn   = lane & 15;           // N (column) inside D tile

  // init: h0 = 0, cache biases / layer-0 input weights
  for (int i = tid; i < MB * ASTR; i += 256) A[i] = 0;
  for (int i = tid; i < G3; i += 256) {
    bih_s[i]  = bih[i];
    bhh_s[i]  = bhh[i];
    wih0_s[i] = (layer == 0) ? wih0[i] : 0.0f;
  }
  __syncthreads();

  for (int t = 0; t < TT; ++t) {
    // ---- phase 0: stage this step's input into the A panel ----
    if (layer == 0) {
      if (tid < MB) {
        int n = base + tid;                       // n = b*32 + c
        xs[tid] = x_in[((size_t)((n >> 5) * TT + t)) * 32 + (n & 31)];
      }
    } else {
      for (int i = tid; i < MB * H; i += 256) {
        int row = i >> 8;
        int k   = i & (H - 1);
        const float* src = x_in + ((size_t)(base + row) * TT + t) * H + k;
        A[row * ASTR + 256 + k] = f2bf(*src);
        if ((k & 31) == 0 && t + 1 < TT)
          __builtin_prefetch(src + H, 0, 0);      // next timestep's row -> global_prefetch_b8
      }
    }
    __syncthreads();

    // ---- phase 1: r,z gates (fused K = h(256) [+ x(256)]) ----
    for (int it = 0; it < P1_ITERS; ++it) {
      int tt   = it * 8 + w;
      int rt   = tt >> 5;           // row tile
      int c    = tt & 31;           // gate col-tile: 0..15 = r, 16..31 = z
      int rowb = rt * 16;
      int arow = rowb + (lane & 15);
      v8f acc = {};
      const unsigned short* bp = whh_p + ((size_t)(c * 8) * 32 + lane) * 16;
#pragma unroll
      for (int kf = 0; kf < 8; ++kf) {
        BfFrag a, b;
        const uint4* ap = (const uint4*)(A + arow * ASTR + kf * 32 + (m_hi << 3));
        a.q[0] = ap[0];  a.q[1] = ap[2];         // K..K+7 and K+16..K+23
        const uint4* bq = (const uint4*)(bp + (size_t)kf * 32 * 16);
        b.q[0] = bq[0];  b.q[1] = bq[1];
        acc = __builtin_amdgcn_wmma_f32_16x16x32_bf16(false, a.v, false, b.v,
                                                      (short)0, acc, false, false);
      }
      if (layer != 0) {
        const unsigned short* bp2 = wih_p + ((size_t)(c * 8) * 32 + lane) * 16;
#pragma unroll
        for (int kf = 0; kf < 8; ++kf) {
          BfFrag a, b;
          const uint4* ap = (const uint4*)(A + arow * ASTR + 256 + kf * 32 + (m_hi << 3));
          a.q[0] = ap[0];  a.q[1] = ap[2];
          const uint4* bq = (const uint4*)(bp2 + (size_t)kf * 32 * 16);
          b.q[0] = bq[0];  b.q[1] = bq[1];
          acc = __builtin_amdgcn_wmma_f32_16x16x32_bf16(false, a.v, false, b.v,
                                                        (short)0, acc, false, false);
        }
      }
#pragma unroll
      for (int e = 0; e < 8; ++e) {
        int row = rowb + e + (m_hi << 3);
        int gc  = c * 16 + nn;                                 // 0..511
        float gi0 = (layer == 0) ? (xs[row] * wih0_s[gc]) : 0.0f;
        float v = acc[e] + gi0 + bih_s[gc] + bhh_s[gc];
        unsigned short sb = f2bf(sigmoidf(v));
        if (gc < H) r_s[row * H + gc]       = sb;
        else        z_s[row * H + (gc - H)] = sb;
      }
    }
    __syncthreads();

    // ---- phase 2a: n gate (gi and gh kept separate) + GRU update ----
    for (int it = 0; it < P2_ITERS; ++it) {
      int tt    = it * 8 + w;
      int rt    = tt >> 4;
      int c2    = tt & 15;          // n-gate col tile
      int rowb  = rt * 16;
      int arow  = rowb + (lane & 15);
      int cglob = 32 + c2;
      v8f acch = {};
      v8f acci = {};
      const unsigned short* bph = whh_p + ((size_t)(cglob * 8) * 32 + lane) * 16;
#pragma unroll
      for (int kf = 0; kf < 8; ++kf) {
        BfFrag a, b;
        const uint4* ap = (const uint4*)(A + arow * ASTR + kf * 32 + (m_hi << 3));
        a.q[0] = ap[0];  a.q[1] = ap[2];
        const uint4* bq = (const uint4*)(bph + (size_t)kf * 32 * 16);
        b.q[0] = bq[0];  b.q[1] = bq[1];
        acch = __builtin_amdgcn_wmma_f32_16x16x32_bf16(false, a.v, false, b.v,
                                                       (short)0, acch, false, false);
      }
      if (layer != 0) {
        const unsigned short* bpi = wih_p + ((size_t)(cglob * 8) * 32 + lane) * 16;
#pragma unroll
        for (int kf = 0; kf < 8; ++kf) {
          BfFrag a, b;
          const uint4* ap = (const uint4*)(A + arow * ASTR + 256 + kf * 32 + (m_hi << 3));
          a.q[0] = ap[0];  a.q[1] = ap[2];
          const uint4* bq = (const uint4*)(bpi + (size_t)kf * 32 * 16);
          b.q[0] = bq[0];  b.q[1] = bq[1];
          acci = __builtin_amdgcn_wmma_f32_16x16x32_bf16(false, a.v, false, b.v,
                                                         (short)0, acci, false, false);
        }
      }
#pragma unroll
      for (int e = 0; e < 8; ++e) {
        int row = rowb + e + (m_hi << 3);
        int hc  = c2 * 16 + nn;                 // hidden col 0..255
        int gc  = 512 + hc;
        float gi = ((layer == 0) ? (xs[row] * wih0_s[gc]) : acci[e]) + bih_s[gc];
        float gh = acch[e] + bhh_s[gc];
        float r    = bf2f(r_s[row * H + hc]);
        float z    = bf2f(z_s[row * H + hc]);
        float hold = bf2f(A[row * ASTR + hc]);
        float n    = tanhf(gi + r * gh);
        float hnew = (1.0f - z) * n + z * hold;
        y_out[((size_t)(base + row) * TT + t) * H + hc] = hnew;
        hn_s[row * H + hc] = f2bf(hnew);
      }
    }
    __syncthreads();

    // ---- phase 2b: commit h_new into the A panel for step t+1 ----
    for (int i = tid; i < MB * H; i += 256) {
      int row = i >> 8;
      int k   = i & (H - 1);
      A[row * ASTR + k] = hn_s[i];
    }
    __syncthreads();
  }
}

extern "C" void kernel_launch(void* const* d_in, const int* in_sizes, int n_in,
                              void* d_out, int out_size, void* d_ws, size_t ws_size,
                              hipStream_t stream) {
  (void)in_sizes; (void)n_in; (void)out_size; (void)ws_size;
  const float* inputs = (const float*)d_in[0];
  const float* Wih[3] = {(const float*)d_in[1], (const float*)d_in[5], (const float*)d_in[9]};
  const float* Whh[3] = {(const float*)d_in[2], (const float*)d_in[6], (const float*)d_in[10]};
  const float* bih[3] = {(const float*)d_in[3], (const float*)d_in[7], (const float*)d_in[11]};
  const float* bhh[3] = {(const float*)d_in[4], (const float*)d_in[8], (const float*)d_in[12]};

  unsigned short* ws = (unsigned short*)d_ws;
  unsigned short* whh_p[3];
  unsigned short* wih_p[3];
  for (int l = 0; l < 3; ++l) {
    whh_p[l] = ws + (size_t)l * PACK_ELEMS;
    wih_p[l] = ws + (size_t)(3 + l) * PACK_ELEMS;
  }

  dim3 pb(256), pg((PACK_ELEMS + 255) / 256);
  for (int l = 0; l < 3; ++l)
    gru_pack_b<<<pg, pb, 0, stream>>>(Whh[l], whh_p[l]);
  for (int l = 1; l < 3; ++l)
    gru_pack_b<<<pg, pb, 0, stream>>>(Wih[l], wih_p[l]);

  float* y = (float*)d_out;
  // Layer outputs are written in place into d_out (read-before-write per WG/step).
  gru_layer_kernel<<<NWG, 256, 0, stream>>>(0, inputs, whh_p[0], whh_p[0], Wih[0], bih[0], bhh[0], y);
  gru_layer_kernel<<<NWG, 256, 0, stream>>>(1, y,      whh_p[1], wih_p[1], Wih[1], bih[1], bhh[1], y);
  gru_layer_kernel<<<NWG, 256, 0, stream>>>(2, y,      whh_p[2], wih_p[2], Wih[2], bih[2], bhh[2], y);
}
